// ParallelText2Mel_12773232738681
// MI455X (gfx1250) — matrix operand order repaired and checked by reading, not compiled, not run on hardware
//
#include <hip/hip_runtime.h>
#include <hip/hip_bf16.h>

// ---------------------------------------------------------------------------
// CDNA5 (gfx1250) WMMA implementation of ParallelText2Mel forward.
// - Conv1d layers: 16x16x32 f16 WMMA GEMMs, f32 accumulate.
// - All inter-layer activations stored f16 (bandwidth-bound workload).
// - Halo tile staged global->LDS via async b128 copies (ASYNCcnt) when the
//   toolchain exposes the gfx1250 builtin; plain b128 copy otherwise.
// - Weights pre-packed to f16 in B-fragment lane order (2x b128 loads/frag).
// - 32 time steps per workgroup: each wave does 2 M-tiles per B fragment.
// ---------------------------------------------------------------------------

typedef _Float16 v16h __attribute__((ext_vector_type(16)));
typedef float    v8f  __attribute__((ext_vector_type(8)));
typedef int      vs4i __attribute__((vector_size(4 * sizeof(int))));

#define BATCH 32
#define SEQ_N 256
#define CH    128
#define EPS_F 1e-5f

union H8  { float4 f4; _Float16 h[8]; };
union V16 { v16h v;    _Float16 h[16]; };

#if __has_builtin(__builtin_amdgcn_global_load_async_to_lds_b128)
#define HAVE_ASYNC_LDS 1
#else
#define HAVE_ASYNC_LDS 0
#endif

__device__ __forceinline__ void async_wait0() {
#if __has_builtin(__builtin_amdgcn_s_wait_asynccnt)
  __builtin_amdgcn_s_wait_asynccnt(0);
#else
  asm volatile("s_wait_asynccnt 0x0" ::: "memory");
#endif
}

// ---- embedding gather (f32 table -> f16 activations) -----------------------
__global__ __launch_bounds__(256)
void embed_kernel(const int* __restrict__ texts, const float* __restrict__ table,
                  _Float16* __restrict__ out, int total) {
  int i = blockIdx.x * 256 + threadIdx.x;
  if (i >= total) return;
  int c  = i & (CH - 1);
  int bn = i >> 7;
  out[i] = (_Float16)table[texts[bn] * CH + c];
}

// ---- weight pre-pack: f32 (k,Cin,Cout) -> f16 in B-fragment lane order -----
// Fragment (nt, j, cg) = 512 halves: index lane*16+e holds
// w[j, cg*32 + (lane>=16?16:0) + e, nt*16 + (lane&15)].
__global__ __launch_bounds__(256)
void pack_w_kernel(const float* __restrict__ w, _Float16* __restrict__ pw,
                   int K, int Cout, int total) {
  int idx = blockIdx.x * 256 + threadIdx.x;
  if (idx >= total) return;
  int frag = idx >> 9;
  int r    = idx & 511;
  int lane = r >> 4;
  int e    = r & 15;
  int cg   = frag & 3;
  int j    = (frag >> 2) % K;
  int nt   = frag / (4 * K);
  int ci   = cg * 32 + ((lane >> 4) ? 16 : 0) + e;
  int col  = nt * 16 + (lane & 15);
  pw[idx]  = (_Float16)w[(size_t)(j * CH + ci) * Cout + col];
}

// ---- WMMA conv1d: y[t] = sum_j sum_ci x[t-b0+j*dil, ci] * w[j,ci,co] + b ---
__global__ __launch_bounds__(256)
void conv_wmma_kernel(const _Float16* __restrict__ x, const _Float16* __restrict__ pw,
                      const float* __restrict__ bias, const _Float16* __restrict__ resid,
                      void* __restrict__ out, float* __restrict__ stats,
                      int T, int K, int DIL, int Cout, int doRelu, int outHalf) {
  __shared__ __align__(16) _Float16 sx[56 * CH];   // 32 + (K-1)*DIL <= 56 rows
  __shared__ float ssum[CH], ssq[CH];

  const int tid  = threadIdx.x;
  const int nT   = (T + 31) >> 5;
  const int b    = blockIdx.x / nT;
  const int t0   = (blockIdx.x % nT) << 5;
  const int tot  = DIL * (K - 1);
  const int b0   = tot >> 1;
  const int rows = 32 + tot;
  const int validEnd = T - (tot - b0);

  // Stage f16 input halo tile into LDS: 16B chunks, async DMA when available.
  const int chunks = rows << 4;                  // 16 x 16B chunks per row
  for (int i = tid; i < chunks; i += 256) {
    int r  = i >> 4;
    int c8 = (i & 15) << 3;                      // 8 halves per chunk
    int t_in = t0 - b0 + r;
    _Float16* lp = &sx[r * CH + c8];
    if (t_in >= 0 && t_in < T) {
      const _Float16* gp = &x[((size_t)b * T + t_in) * CH + c8];
#if HAVE_ASYNC_LDS
      __builtin_amdgcn_global_load_async_to_lds_b128(
          (__attribute__((address_space(1))) vs4i*)gp,
          (__attribute__((address_space(3))) vs4i*)lp, 0, 0);
#else
      *reinterpret_cast<float4*>(lp) = *reinterpret_cast<const float4*>(gp);
#endif
    } else {
      float4 z = {0.f, 0.f, 0.f, 0.f};
      *reinterpret_cast<float4*>(lp) = z;
    }
  }
#if HAVE_ASYNC_LDS
  async_wait0();
#endif
  if (stats && tid < CH) { ssum[tid] = 0.f; ssq[tid] = 0.f; }
  __syncthreads();

  const int wave = tid >> 5;
  const int lane = tid & 31;
  const int hi   = lane >> 4;     // K-half select (contiguous k = 16*hi + e)
  const int lrow = lane & 15;

  for (int nt = wave; nt * 16 < Cout; nt += 8) {   // wave-uniform loop
    const int ncol = nt * 16 + lrow;
    v8f acc[2];
    acc[0] = (v8f){0.f,0.f,0.f,0.f,0.f,0.f,0.f,0.f};
    acc[1] = (v8f){0.f,0.f,0.f,0.f,0.f,0.f,0.f,0.f};
    for (int j = 0; j < K; ++j) {
      #pragma unroll
      for (int cg = 0; cg < 4; ++cg) {             // 32-channel K chunks
        // B fragment (32x16 f16): two 16B global loads from packed weights.
        const _Float16* fp = pw + ((((size_t)nt * K + j) * 4 + cg) << 9) + (lane << 4);
        H8 bl, bh;
        bl.f4 = *reinterpret_cast<const float4*>(fp);
        bh.f4 = *reinterpret_cast<const float4*>(fp + 8);
        V16 Bv;
        #pragma unroll
        for (int e = 0; e < 8; ++e) { Bv.h[e] = bl.h[e]; Bv.h[e + 8] = bh.h[e]; }
        // Two M-tiles share this B fragment.
        #pragma unroll
        for (int mt = 0; mt < 2; ++mt) {
          int abase = (mt * 16 + lrow + j * DIL) * CH + cg * 32 + hi * 16;
          H8 al, ah;
          al.f4 = *reinterpret_cast<const float4*>(&sx[abase]);
          ah.f4 = *reinterpret_cast<const float4*>(&sx[abase + 8]);
          V16 A;
          #pragma unroll
          for (int e = 0; e < 8; ++e) { A.h[e] = al.h[e]; A.h[e + 8] = ah.h[e]; }
          acc[mt] = __builtin_amdgcn_wmma_f32_16x16x32_f16(false, A.v, false, Bv.v,
                                                           (short)0, acc[mt], false, false);
        }
      }
    }
    // Epilogue: bias + pad-mask + relu + residual + store + stats.
    float lsum = 0.f, lsq = 0.f;
    #pragma unroll
    for (int mt = 0; mt < 2; ++mt) {
      #pragma unroll
      for (int r = 0; r < 8; ++r) {
        int t = t0 + mt * 16 + r + hi * 8;    // D layout: M split at lane 16
        if (t >= T) continue;
        float v = acc[mt][r] + bias[ncol];
        if (t < b0 || t >= validEnd) v = 0.f;   // ZeroTemporalPad
        if (doRelu) v = fmaxf(v, 0.f);
        size_t oi = ((size_t)b * T + t) * Cout + ncol;
        if (resid) v += (float)resid[oi];
        if (outHalf) ((_Float16*)out)[oi] = (_Float16)v;
        else         ((float*)out)[oi]    = v;
        lsum += v; lsq += v * v;
      }
    }
    if (stats) { atomicAdd(&ssum[ncol], lsum); atomicAdd(&ssq[ncol], lsq); }
  }
  if (stats) {
    __syncthreads();
    if (tid < CH) {
      atomicAdd(&stats[tid], ssum[tid]);
      atomicAdd(&stats[CH + tid], ssq[tid]);
    }
  }
}

// ---- freqnorm apply: out(f16) = [resid +] (relu?(x) - mean)*rsqrt(var+eps) -
__global__ __launch_bounds__(256)
void norm_kernel(const float* __restrict__ tmp, const float* __restrict__ stats,
                 const _Float16* __restrict__ resid, _Float16* __restrict__ out,
                 int total, float invN, int doRelu) {
  int i = blockIdx.x * 256 + threadIdx.x;
  if (i >= total) return;
  int c = i & (CH - 1);
  float m = stats[c] * invN;
  float v = stats[CH + c] * invN - m * m;
  float x = tmp[i];
  if (doRelu) x = fmaxf(x, 0.f);
  float y = (x - m) * rsqrtf(v + EPS_F);
  if (resid) y += (float)resid[i];
  out[i] = (_Float16)y;
}

// ---- standalone relu-stats (for enc_post path, f32 input) ------------------
__global__ __launch_bounds__(256)
void stats_relu_kernel(const float* __restrict__ x, float* __restrict__ stats, int total) {
  __shared__ float ssum[CH], ssq[CH];
  int tid = threadIdx.x;
  if (tid < CH) { ssum[tid] = 0.f; ssq[tid] = 0.f; }
  __syncthreads();
  int i = blockIdx.x * 256 + tid;
  if (i < total) {
    float v = fmaxf(x[i], 0.f);
    int c = i & (CH - 1);
    atomicAdd(&ssum[c], v);
    atomicAdd(&ssq[c], v * v);
  }
  __syncthreads();
  if (tid < CH) {
    atomicAdd(&stats[tid], ssum[tid]);
    atomicAdd(&stats[CH + tid], ssq[tid]);
  }
}

// ---- duration cumsum -------------------------------------------------------
__global__ void cumsum_kernel(const int* __restrict__ dur, int* __restrict__ cums) {
  int b = blockIdx.x;
  if (threadIdx.x == 0) {
    int s = 0;
    for (int n = 0; n < SEQ_N; ++n) { s += dur[b * SEQ_N + n]; cums[b * SEQ_N + n] = s; }
  }
}

// ---- length regulator: gather + positional encoding + validity mask --------
__global__ __launch_bounds__(128)
void regulate_kernel(const _Float16* __restrict__ enc, const int* __restrict__ cums,
                     _Float16* __restrict__ out, int T) {
  int b = blockIdx.x / T;
  int t = blockIdx.x % T;
  __shared__ int sidx, svalid;
  if (threadIdx.x == 0) {
    const int* c = &cums[b * SEQ_N];
    int lo = 0, hi2 = SEQ_N;                     // searchsorted(side='right')
    while (lo < hi2) { int mid = (lo + hi2) >> 1; if (c[mid] <= t) lo = mid + 1; else hi2 = mid; }
    sidx   = min(lo, SEQ_N - 1);
    svalid = (t < c[SEQ_N - 1]) ? 1 : 0;
  }
  __syncthreads();
  int ch = threadIdx.x;
  float v = 0.f;
  if (svalid) {
    int i2 = ch >> 1;
    float ang = (float)t * __expf(-logf(10000.f) * (2.f * (float)i2) / (float)CH);
    float pe  = (ch & 1) ? __cosf(ang) : __sinf(ang);
    v = (float)enc[((size_t)b * SEQ_N + sidx) * CH + ch] + pe;
  }
  out[((size_t)b * T + t) * CH + ch] = (_Float16)v;
}

// ---- duration head: C -> 1 dot product (f32 output) ------------------------
__global__ __launch_bounds__(128)
void durout_kernel(const _Float16* __restrict__ x, const float* __restrict__ w,
                   const float* __restrict__ bias, float* __restrict__ out) {
  __shared__ float red[128];
  int bn = blockIdx.x;
  int c  = threadIdx.x;
  red[c] = (float)x[(size_t)bn * CH + c] * w[c];
  __syncthreads();
  for (int s = 64; s > 0; s >>= 1) { if (c < s) red[c] += red[c + s]; __syncthreads(); }
  if (c == 0) out[bn] = red[0] + bias[0];
}

// ---------------------------------------------------------------------------
extern "C" void kernel_launch(void* const* d_in, const int* in_sizes, int n_in,
                              void* d_out, int out_size, void* d_ws, size_t ws_size,
                              hipStream_t stream) {
  (void)in_sizes; (void)n_in; (void)ws_size;
  const int B = BATCH, N = SEQ_N, C = CH, OUTC = 80;
  const int T = (out_size - B * N) / (B * OUTC);   // recover T_out on host
  if (T <= 0) return;

  const int* texts = (const int*)d_in[0];
  const int* durs  = (const int*)d_in[2];

  // Param pointers (setup_inputs dict insertion order, nested lists in order).
  const float* p_embed  = (const float*)d_in[4];
  const float* prenet_w = (const float*)d_in[5];
  const float* prenet_b = (const float*)d_in[6];
  auto encW = [&](int blk, int cv) { return (const float*)d_in[7 + (blk * 2 + cv) * 2]; };
  auto encB = [&](int blk, int cv) { return (const float*)d_in[8 + (blk * 2 + cv) * 2]; };
  const float* post1_w = (const float*)d_in[47]; const float* post1_b = (const float*)d_in[48];
  const float* post2_w = (const float*)d_in[49]; const float* post2_b = (const float*)d_in[50];
  auto durW = [&](int i) { return (const float*)d_in[51 + i * 2]; };
  auto durB = [&](int i) { return (const float*)d_in[52 + i * 2]; };
  const float* durout_w = (const float*)d_in[57]; const float* durout_b = (const float*)d_in[58];
  auto decW = [&](int blk, int cv) { return (const float*)d_in[59 + (blk * 2 + cv) * 2]; };
  auto decB = [&](int blk, int cv) { return (const float*)d_in[60 + (blk * 2 + cv) * 2]; };
  const float* dp1_w = (const float*)d_in[95]; const float* dp1_b = (const float*)d_in[96];
  auto dp2W = [&](int cv) { return (const float*)d_in[97 + cv * 2]; };
  auto dp2B = [&](int cv) { return (const float*)d_in[98 + cv * 2]; };
  const float* dow = (const float*)d_in[101]; const float* dob = (const float*)d_in[102];

  float* mel = (float*)d_out;
  float* prd = (float*)d_out + (size_t)B * T * OUTC;

  // Workspace carve.
  char* wp8 = (char*)d_ws;
  auto carve = [&](size_t bytes) -> void* {
    void* p = wp8; wp8 += (bytes + 255) & ~(size_t)255; return p;
  };
  float* stats = (float*)carve(2 * C * sizeof(float));
  int*   cums  = (int*)carve((size_t)B * N * sizeof(int));
  size_t SE = (size_t)B * N * C, SD = (size_t)B * T * C;
  // f16 activation buffers
  _Float16* aH   = (_Float16*)carve(SE * 2);
  _Float16* hH   = (_Float16*)carve(SE * 2);
  _Float16* xH   = (_Float16*)carve(SE * 2);
  _Float16* encH = (_Float16*)carve(SE * 2);
  _Float16* dExp = (_Float16*)carve(SD * 2);
  _Float16* dXH  = (_Float16*)carve(SD * 2);
  _Float16* dYH  = (_Float16*)carve(SD * 2);
  // f32 conv-output temporaries (freqnorm statistics path)
  float* eT = (float*)carve(SE * 4);
  float* dT = (float*)carve(SD * 4);

  // Pre-pack all conv weights to f16 fragment order.
  auto packW = [&](const float* w, int K, int Cout) -> const _Float16* {
    int total = (Cout / 16) * K * 4 * 512;       // == K*128*Cout halves
    _Float16* pw = (_Float16*)carve((size_t)total * sizeof(_Float16));
    pack_w_kernel<<<(total + 255) / 256, 256, 0, stream>>>(w, pw, K, Cout, total);
    return pw;
  };
  const _Float16* PWpre = packW(prenet_w, 1, C);
  const _Float16* PWenc[10][2];
  for (int i = 0; i < 10; ++i)
    for (int cv = 0; cv < 2; ++cv) PWenc[i][cv] = packW(encW(i, cv), 4, C);
  const _Float16* PWp1 = packW(post1_w, 1, C);
  const _Float16* PWp2 = packW(post2_w, 1, C);
  static const int DUR_K[3] = {4, 3, 1};
  const _Float16* PWdur[3];
  for (int i = 0; i < 3; ++i) PWdur[i] = packW(durW(i), DUR_K[i], C);
  const _Float16* PWdec[9][2];
  for (int i = 0; i < 9; ++i)
    for (int cv = 0; cv < 2; ++cv) PWdec[i][cv] = packW(decW(i, cv), 4, C);
  const _Float16* PWdp1 = packW(dp1_w, 1, C);
  const _Float16* PWdp2[2] = { packW(dp2W(0), 4, C), packW(dp2W(1), 4, C) };
  const _Float16* PWout = packW(dow, 1, OUTC);

  const float invE = 1.f / ((float)B * N);
  const float invD = 1.f / ((float)B * T);

  auto conv = [&](const _Float16* x, const _Float16* pw, const float* bias,
                  const _Float16* resid, void* out, int outHalf, float* st,
                  int Tlen, int Kk, int dil, int Cout, int relu) {
    int grid = B * ((Tlen + 31) >> 5);
    conv_wmma_kernel<<<grid, 256, 0, stream>>>(x, pw, bias, resid, out, st,
                                               Tlen, Kk, dil, Cout, relu, outHalf);
  };
  auto norm = [&](const float* tmp, const _Float16* resid, _Float16* out, size_t total,
                  float invN, int relu) {
    norm_kernel<<<(int)((total + 255) >> 8), 256, 0, stream>>>(tmp, stats, resid, out,
                                                               (int)total, invN, relu);
  };
  auto clearStats = [&]() { (void)hipMemsetAsync(stats, 0, 2 * C * sizeof(float), stream); };

  static const int ENC_DIL[10] = {1, 2, 4, 1, 2, 4, 1, 2, 4, 1};
  static const int DEC_DIL[9]  = {1, 2, 4, 8, 1, 2, 4, 8, 1};

  // ---- TextEncoder ----
  embed_kernel<<<(int)((SE + 255) >> 8), 256, 0, stream>>>(texts, p_embed, aH, (int)SE);
  conv(aH, PWpre, prenet_b, nullptr, hH, 1, nullptr, N, 1, 1, C, 1);   // h = relu(prenet)
  const _Float16* xin = hH;
  for (int i = 0; i < 10; ++i) {
    int d = ENC_DIL[i];
    clearStats(); conv(xin, PWenc[i][0], encB(i, 0), nullptr, eT, 0, stats, N, 4, d, C, 1);
    norm(eT, nullptr, aH, SE, invE, 0);
    clearStats(); conv(aH, PWenc[i][1], encB(i, 1), nullptr, eT, 0, stats, N, 4, d, C, 1);
    norm(eT, xin, xH, SE, invE, 0);
    xin = xH;
  }
  conv(xH, PWp1, post1_b, hH, eT, 0, nullptr, N, 1, 1, C, 0);          // x = post1(x)+h
  clearStats();
  stats_relu_kernel<<<(int)((SE + 255) >> 8), 256, 0, stream>>>(eT, stats, (int)SE);
  norm(eT, nullptr, aH, SE, invE, 1);                                  // freqnorm(relu(x))
  conv(aH, PWp2, post2_b, nullptr, encH, 1, nullptr, N, 1, 1, C, 0);   // enc

  // ---- DurationPredictor ----
  xin = encH;
  _Float16* durbuf[2] = { xH, aH };
  for (int i = 0; i < 3; ++i) {
    clearStats(); conv(xin, PWdur[i], durB(i), nullptr, eT, 0, stats, N, DUR_K[i], 1, C, 1);
    _Float16* o = durbuf[i & 1];
    norm(eT, xin, o, SE, invE, 0);
    xin = o;
  }
  durout_kernel<<<B * N, 128, 0, stream>>>(xin, durout_w, durout_b, prd);

  // ---- Length regulator ----
  cumsum_kernel<<<B, 32, 0, stream>>>(durs, cums);
  regulate_kernel<<<B * T, 128, 0, stream>>>(encH, cums, dExp, T);

  // ---- SpecDecoder ----
  xin = dExp;
  for (int i = 0; i < 9; ++i) {
    int d = DEC_DIL[i];
    clearStats(); conv(xin, PWdec[i][0], decB(i, 0), nullptr, dT, 0, stats, T, 4, d, C, 1);
    norm(dT, nullptr, dYH, SD, invD, 0);
    clearStats(); conv(dYH, PWdec[i][1], decB(i, 1), nullptr, dT, 0, stats, T, 4, d, C, 1);
    norm(dT, xin, dXH, SD, invD, 0);
    xin = dXH;
  }
  conv(dXH, PWdp1, dp1_b, dExp, dYH, 1, nullptr, T, 1, 1, C, 0);       // z = post1 + exp
  clearStats(); conv(dYH, PWdp2[0], dp2B(0), nullptr, dT, 0, stats, T, 4, 1, C, 1);
  norm(dT, nullptr, dXH, SD, invD, 0);
  clearStats(); conv(dXH, PWdp2[1], dp2B(1), nullptr, dT, 0, stats, T, 4, 1, C, 1);
  norm(dT, dYH, dXH, SD, invD, 0);
  conv(dXH, PWout, dob, nullptr, mel, 0, nullptr, T, 1, 1, OUTC, 0);   // mel (B,T,80)
}
